// LSTMDecoder_30348238913922
// MI455X (gfx1250) — matrix operand order repaired and checked by reading, not compiled, not run on hardware
//
#include <hip/hip_runtime.h>
#include <math.h>
#include <stddef.h>

typedef __attribute__((ext_vector_type(16))) _Float16 v16h;
typedef __attribute__((ext_vector_type(8)))  _Float16 v8h;
typedef __attribute__((ext_vector_type(8)))  float    v8f;

#define BB 8
#define LL 160
#define TT 256
#define DD 256
#define HH 256
#define AA 256
#define VV 500
#define G4H 1024
#define ML (BB*LL)   /* 1280 */
#define MT (BB*TT)   /* 2048 */

// ---------------------------------------------------------------------------
// WMMA helpers (CDNA5 gfx1250, wave32, 16x16x32 f16 -> f32)
// ---------------------------------------------------------------------------
__device__ __forceinline__ v8f wmma32(v16h a, v16h b, v8f c) {
  return __builtin_amdgcn_wmma_f32_16x16x32_f16(
      /*neg_a=*/false, a, /*neg_b=*/false, b,
      /*c_mod=*/(short)0, c, /*reuse_a=*/false, /*reuse_b=*/false);
}

// A matrix 16x32 f16: lane half h=lane/16, row m chosen by caller.
// per lane: one contiguous 16-half chunk at k0+16h.
__device__ __forceinline__ v16h load_a_frag(const _Float16* rowptr, int k0, int lane) {
  const int h = (lane >> 4) & 1;
  const _Float16* p = rowptr + k0 + 16 * h;
  v16h r = *(const v16h*)(p);
  return r;
}

// B matrix 32x16 f16 where B[k][n] = W[n][k] (W stored (N,K) row-major):
// per lane: one contiguous 16-half chunk of row n of W.
__device__ __forceinline__ v16h load_b_frag(const _Float16* W, int ldw, int n0,
                                            int k0, int lane) {
  const int n = n0 + (lane & 15);
  const int koff = k0 + ((lane >> 4) & 1) * 16;
  return *(const v16h*)(W + (size_t)n * ldw + koff);
}

__device__ __forceinline__ float sigmf(float x) { return 1.f / (1.f + __expf(-x)); }

// ---------------------------------------------------------------------------
// Prep kernels: f32 -> f16 conversions / padding / transpose
// ---------------------------------------------------------------------------
__global__ void k_f32_to_f16(const float* __restrict__ src, _Float16* __restrict__ dst, int n) {
  int i = blockIdx.x * blockDim.x + threadIdx.x;
  if (i < n) dst[i] = (_Float16)src[i];
}

__global__ void k_outw_pad(const float* __restrict__ src, _Float16* __restrict__ dst) {
  int i = blockIdx.x * blockDim.x + threadIdx.x;  // 512*512
  int r = i >> 9, c = i & 511;
  dst[i] = (_Float16)((r < VV) ? src[r * 512 + c] : 0.f);
}

__global__ void k_mem_prep(const float* __restrict__ mem, _Float16* __restrict__ memf,
                           _Float16* __restrict__ memT) {
  int i = blockIdx.x * blockDim.x + threadIdx.x;  // B*T*D = 524288
  int b = i >> 16, t = (i >> 8) & 255, d = i & 255;
  float v = mem[i];
  memf[i] = (_Float16)v;
  memT[((size_t)b << 16) + (d << 8) + t] = (_Float16)v;
}

// ---------------------------------------------------------------------------
// x_proj = embed[ids] @ W_ih^T + b_ih + b_hh   (M=1280, N=1024, K=256)
// ---------------------------------------------------------------------------
__global__ void __launch_bounds__(256)
k_xproj(const int* __restrict__ ids, const _Float16* __restrict__ embf,
        const _Float16* __restrict__ Wih, const float* __restrict__ bih,
        const float* __restrict__ bhh, float* __restrict__ xproj) {
  const int lane = threadIdx.x & 31;
  const int gw = blockIdx.x * 8 + (threadIdx.x >> 5);  // 5120 wave-tiles
  const int tm = gw >> 6;                              // 80 M-tiles
  const int tn = gw & 63;                              // 64 N-tiles
  const int m = tm * 16 + (lane & 15);
  const _Float16* arow = embf + (size_t)ids[m] * 256;  // embedding gather fused
  const int n0 = tn * 16;
  v8f acc = {};
  for (int k0 = 0; k0 < 256; k0 += 32)
    acc = wmma32(load_a_frag(arow, k0, lane), load_b_frag(Wih, 256, n0, k0, lane), acc);
  const int half = lane >> 4, n = n0 + (lane & 15);
  const float bias = bih[n] + bhh[n];
#pragma unroll
  for (int v = 0; v < 8; ++v)
    xproj[(size_t)(tm * 16 + 8 * half + v) * G4H + n] = acc[v] + bias;
}

// ---------------------------------------------------------------------------
// LSTM scan: single workgroup (32 waves). Per step: async-copy the x_proj
// slice for step l into LDS (ASYNCcnt path) overlapped with the h@Whh^T WMMA
// GEMM; then gates are formed in place in LDS and the cell update runs.
// ---------------------------------------------------------------------------
struct SmemLSTM {
  _Float16 hbuf[16 * HH];   // 8KB, rows 8..15 stay zero (M padding)
  float    cbuf[BB * HH];   // 8KB
  float    xbuf[BB * G4H];  // 32KB: async-load dest, then gates in place
};

__global__ void __launch_bounds__(1024)
k_lstm(const float* __restrict__ xproj, const _Float16* __restrict__ Whh,
       float* __restrict__ outs32, _Float16* __restrict__ outs16) {
  __shared__ __align__(16) SmemLSTM s;
  const int tid = threadIdx.x;
  for (int i = tid; i < 16 * HH; i += 1024) s.hbuf[i] = (_Float16)0.f;
  for (int i = tid; i < BB * HH; i += 1024) s.cbuf[i] = 0.f;
  __syncthreads();

  const int wave = tid >> 5, lane = tid & 31;
  const int mrow = lane & 15, half = lane >> 4;
  // LDS byte offset of xbuf (workgroup's LDS allocation assumed at offset 0)
  const unsigned xbase = (unsigned)offsetof(SmemLSTM, xbuf);

  for (int l = 0; l < LL; ++l) {
    // (a) issue async global->LDS copy of x_proj slice l (2 x b128 per thread)
#pragma unroll
    for (int c = 0; c < 2; ++c) {
      const int e = tid * 4 + c * 4096;        // covers 8192 floats
      const int m = e >> 10, n = e & 1023;
      const float* src = xproj + ((size_t)(m * LL + l) * G4H + n);
      const unsigned ldst = xbase + (unsigned)e * 4u;
      asm volatile("global_load_async_to_lds_b128 %0, %1, off"
                   :: "v"(ldst), "v"(src) : "memory");
    }

    // (b) recurrent GEMM (overlaps with the async copy): wave handles 2 N-tiles
    v8f acc0 = {}, acc1 = {};
    const int n0a = (wave * 2) * 16, n0b = (wave * 2 + 1) * 16;
    for (int k0 = 0; k0 < HH; k0 += 32) {
      v16h a = load_a_frag(s.hbuf + mrow * HH, k0, lane);  // A from LDS
      acc0 = wmma32(a, load_b_frag(Whh, HH, n0a, k0, lane), acc0);
      acc1 = wmma32(a, load_b_frag(Whh, HH, n0b, k0, lane), acc1);
    }

    // (c) wait own async loads, then block-wide barrier so all slices landed
    asm volatile("s_wait_asynccnt 0x0" ::: "memory");
    __syncthreads();

    // (d) gates in place: xbuf[m][n] = xproj_slice + (h @ Whh^T)
    {
      const int na = n0a + (lane & 15), nb = n0b + (lane & 15);
#pragma unroll
      for (int v = 0; v < 8; ++v) {
        const int m = 8 * half + v;
        if (m < BB) {
          s.xbuf[m * G4H + na] += acc0[v];
          s.xbuf[m * G4H + nb] += acc1[v];
        }
      }
    }
    __syncthreads();

    // (e) elementwise cell update: 8*256 cells, 2 per thread
    for (int i = tid; i < BB * HH; i += 1024) {
      const int m = i >> 8, j = i & 255;
      const float ig = s.xbuf[m * G4H + j];
      const float fg = s.xbuf[m * G4H + 256 + j];
      const float gg = s.xbuf[m * G4H + 512 + j];
      const float og = s.xbuf[m * G4H + 768 + j];
      const float c = sigmf(fg) * s.cbuf[i] + sigmf(ig) * tanhf(gg);
      const float h = sigmf(og) * tanhf(c);
      s.cbuf[i] = c;
      s.hbuf[m * HH + j] = (_Float16)h;
      const size_t o = ((size_t)m * LL + l) * HH + j;
      outs32[o] = h;
      outs16[o] = (_Float16)h;
    }
    __syncthreads();
  }
}

// ---------------------------------------------------------------------------
// Generic C(MxN) = A_f16(MxK) @ W_f16(N,K)^T  (f32 out), exact tiling
// ---------------------------------------------------------------------------
__global__ void __launch_bounds__(256)
k_gemm_nt(const _Float16* __restrict__ A, const _Float16* __restrict__ W,
          float* __restrict__ C, int K, int tilesN) {
  const int lane = threadIdx.x & 31;
  const int gw = (blockIdx.x * 256 + threadIdx.x) >> 5;
  const int tm = gw / tilesN, tn = gw % tilesN;
  const int N = tilesN * 16;
  const _Float16* arow = A + (size_t)(tm * 16 + (lane & 15)) * K;
  v8f acc = {};
  for (int k0 = 0; k0 < K; k0 += 32)
    acc = wmma32(load_a_frag(arow, k0, lane), load_b_frag(W, K, tn * 16, k0, lane), acc);
  const int half = lane >> 4, n = tn * 16 + (lane & 15);
#pragma unroll
  for (int v = 0; v < 8; ++v)
    C[(size_t)(tm * 16 + 8 * half + v) * N + n] = acc[v];
}

// ---------------------------------------------------------------------------
// attention scores + softmax: one block per (b,l); e[t] = sum_a v*tanh(q+k)
// ---------------------------------------------------------------------------
__global__ void __launch_bounds__(256)
k_attn(const float* __restrict__ q, const float* __restrict__ kb,
       const float* __restrict__ vw, const unsigned char* __restrict__ mask,
       _Float16* __restrict__ attn) {
  __shared__ float qs[AA];
  __shared__ float vs[AA];
  __shared__ float red[TT];
  const int bl = blockIdx.x;
  const int b = bl / LL;
  const int t = threadIdx.x;
  qs[t] = q[(size_t)bl * AA + t];
  vs[t] = vw[t];
  __syncthreads();
  float e = -INFINITY;
  if (mask[b * TT + t]) {
    const float* kr = kb + ((size_t)b * TT + t) * AA;
    float ssum = 0.f;
#pragma unroll 4
    for (int a = 0; a < AA; ++a) ssum += vs[a] * tanhf(qs[a] + kr[a]);
    e = ssum;
  }
  red[t] = e;
  __syncthreads();
  for (int off = 128; off > 0; off >>= 1) {
    if (t < off) red[t] = fmaxf(red[t], red[t + off]);
    __syncthreads();
  }
  const float mx = red[0];
  __syncthreads();
  const float p = (e == -INFINITY) ? 0.f : __expf(e - mx);
  red[t] = p;
  __syncthreads();
  for (int off = 128; off > 0; off >>= 1) {
    if (t < off) red[t] += red[t + off];
    __syncthreads();
  }
  attn[(size_t)bl * TT + t] = (_Float16)(p / red[0]);
}

// ---------------------------------------------------------------------------
// ctx(1280x256) = attn @ memory[b] ; B operand = memT[b] (D,T) row-major
// ---------------------------------------------------------------------------
__global__ void __launch_bounds__(256)
k_ctx(const _Float16* __restrict__ attn, const _Float16* __restrict__ memT,
      _Float16* __restrict__ ctx) {
  const int lane = threadIdx.x & 31;
  const int gw = (blockIdx.x * 256 + threadIdx.x) >> 5;
  const int tm = gw >> 4, tn = gw & 15;     // 80 x 16 tiles
  const int b = (tm * 16) / LL;             // L=160 multiple of 16 -> uniform
  const _Float16* arow = attn + (size_t)(tm * 16 + (lane & 15)) * TT;
  const _Float16* W = memT + ((size_t)b << 16);
  v8f acc = {};
  for (int k0 = 0; k0 < TT; k0 += 32)
    acc = wmma32(load_a_frag(arow, k0, lane), load_b_frag(W, TT, tn * 16, k0, lane), acc);
  const int half = lane >> 4, n = tn * 16 + (lane & 15);
#pragma unroll
  for (int v = 0; v < 8; ++v)
    ctx[(size_t)(tm * 16 + 8 * half + v) * DD + n] = (_Float16)acc[v];
}

// ---------------------------------------------------------------------------
// logits = [outputs, ctx] @ out_W^T + out_b  (M=1280, K=512, N=500 pad 512)
// ---------------------------------------------------------------------------
__global__ void __launch_bounds__(256)
k_logits(const _Float16* __restrict__ hf, const _Float16* __restrict__ cf,
         const _Float16* __restrict__ Wp, const float* __restrict__ bias,
         float* __restrict__ out) {
  const int lane = threadIdx.x & 31;
  const int gw = (blockIdx.x * 256 + threadIdx.x) >> 5;
  const int tm = gw >> 5, tn = gw & 31;     // 80 x 32 tiles
  const _Float16* hrow = hf + (size_t)(tm * 16 + (lane & 15)) * HH;
  const _Float16* crow = cf + (size_t)(tm * 16 + (lane & 15)) * DD;
  v8f acc = {};
  for (int k0 = 0; k0 < 512; k0 += 32) {
    v16h a = (k0 < 256) ? load_a_frag(hrow, k0, lane) : load_a_frag(crow, k0 - 256, lane);
    acc = wmma32(a, load_b_frag(Wp, 512, tn * 16, k0, lane), acc);
  }
  const int half = lane >> 4, n = tn * 16 + (lane & 15);
  if (n < VV) {
    const float bb = bias[n];
#pragma unroll
    for (int v = 0; v < 8; ++v)
      out[(size_t)(tm * 16 + 8 * half + v) * VV + n] = acc[v] + bb;
  }
}

// ---------------------------------------------------------------------------
extern "C" void kernel_launch(void* const* d_in, const int* in_sizes, int n_in,
                              void* d_out, int out_size, void* d_ws, size_t ws_size,
                              hipStream_t stream) {
  const int* ids            = (const int*)d_in[0];
  const float* memory       = (const float*)d_in[1];
  const unsigned char* mask = (const unsigned char*)d_in[2];
  const float* embed        = (const float*)d_in[3];
  const float* W_ih         = (const float*)d_in[4];
  const float* W_hh         = (const float*)d_in[5];
  const float* b_ih         = (const float*)d_in[6];
  const float* b_hh         = (const float*)d_in[7];
  const float* w_h          = (const float*)d_in[8];
  const float* w_m          = (const float*)d_in[9];
  const float* v_w          = (const float*)d_in[10];
  const float* out_W        = (const float*)d_in[11];
  const float* out_b        = (const float*)d_in[12];
  float* logits             = (float*)d_out;

  char* cur = (char*)d_ws;
  auto alloc = [&](size_t bytes) -> char* {
    char* r = cur;
    cur += (bytes + 255) & ~(size_t)255;
    return r;
  };
  _Float16* embf   = (_Float16*)alloc((size_t)VV * 256 * 2);
  _Float16* wih16  = (_Float16*)alloc((size_t)G4H * 256 * 2);
  _Float16* whh16  = (_Float16*)alloc((size_t)G4H * 256 * 2);
  _Float16* wh16   = (_Float16*)alloc((size_t)AA * HH * 2);
  _Float16* wm16   = (_Float16*)alloc((size_t)AA * DD * 2);
  _Float16* outw16 = (_Float16*)alloc((size_t)512 * 512 * 2);
  _Float16* memf16 = (_Float16*)alloc((size_t)MT * DD * 2);
  _Float16* memT16 = (_Float16*)alloc((size_t)BB * DD * TT * 2);
  _Float16* outs16 = (_Float16*)alloc((size_t)ML * HH * 2);
  _Float16* attn16 = (_Float16*)alloc((size_t)ML * TT * 2);
  _Float16* ctx16  = (_Float16*)alloc((size_t)ML * DD * 2);
  float* xproj     = (float*)alloc((size_t)ML * G4H * 4);
  float* outs32    = (float*)alloc((size_t)ML * HH * 4);
  float* qbuf      = (float*)alloc((size_t)ML * AA * 4);
  float* kbuf      = (float*)alloc((size_t)MT * AA * 4);

  // --- prep: f16 conversions
  k_f32_to_f16<<<(VV * 256 + 255) / 256, 256, 0, stream>>>(embed, embf, VV * 256);
  k_f32_to_f16<<<(G4H * 256 + 255) / 256, 256, 0, stream>>>(W_ih, wih16, G4H * 256);
  k_f32_to_f16<<<(G4H * 256 + 255) / 256, 256, 0, stream>>>(W_hh, whh16, G4H * 256);
  k_f32_to_f16<<<(AA * HH + 255) / 256, 256, 0, stream>>>(w_h, wh16, AA * HH);
  k_f32_to_f16<<<(AA * DD + 255) / 256, 256, 0, stream>>>(w_m, wm16, AA * DD);
  k_outw_pad<<<(512 * 512) / 256, 256, 0, stream>>>(out_W, outw16);
  k_mem_prep<<<(MT * DD) / 256, 256, 0, stream>>>(memory, memf16, memT16);

  // --- x_proj GEMM: 80*64 wave-tiles, 8 waves/block
  k_xproj<<<640, 256, 0, stream>>>(ids, embf, wih16, b_ih, b_hh, xproj);

  // --- LSTM scan: one persistent workgroup (async LDS prefetch + WMMA)
  k_lstm<<<1, 1024, 0, stream>>>(xproj, whh16, outs32, outs16);

  // --- q = outputs @ w_h^T   (80*16 = 1280 wave-tiles)
  k_gemm_nt<<<160, 256, 0, stream>>>(outs16, wh16, qbuf, HH, 16);
  // --- k = memory @ w_m^T    (128*16 = 2048 wave-tiles)
  k_gemm_nt<<<256, 256, 0, stream>>>(memf16, wm16, kbuf, DD, 16);

  // --- additive attention scores + softmax
  k_attn<<<ML, 256, 0, stream>>>(qbuf, kbuf, v_w, mask, attn16);

  // --- ctx = attn @ memory
  k_ctx<<<160, 256, 0, stream>>>(attn16, memT16, ctx16);

  // --- logits = [outputs, ctx] @ out_W^T + out_b
  k_logits<<<320, 256, 0, stream>>>(outs16, ctx16, outw16, out_b, logits);
}
